// MSLoRA_MHA_62156766707798
// MI455X (gfx1250) — matrix operand order repaired and checked by reading, not compiled, not run on hardware
//
#include <hip/hip_runtime.h>
#include <hip/hip_bf16.h>

typedef __bf16 bf16_t;
typedef __attribute__((ext_vector_type(16))) __bf16 v16bf;
typedef __attribute__((ext_vector_type(8)))  float  v8f;

// ---------------------------------------------------------------------------
// WMMA wrapper: D(f32 16x16) = A(bf16 16x32) * B(bf16 32x16) + C
// ---------------------------------------------------------------------------
__device__ __forceinline__ v8f wmma_bf16(v16bf a, v16bf b, v8f c) {
  return __builtin_amdgcn_wmma_f32_16x16x32_bf16(
      /*neg_a=*/false, a, /*neg_b=*/false, b,
      /*c_mod=*/(short)0, c, /*reuse_a=*/false, /*reuse_b=*/false);
}

// A-fragment (16x32 bf16) from bf16 memory, row-major [16 rows][stride halves]
// Lane layout: row = lane&15; lanes 0-15 hold K = 0..15,
// lanes 16-31 hold K = 16..31  (one 32B contiguous K-run per lane).
__device__ __forceinline__ v16bf load_a_bf16(const bf16_t* base, int stride) {
  const int lane = threadIdx.x & 31;
  const bf16_t* p = base + (size_t)(lane & 15) * stride + ((lane >> 4) << 4);
  union { v16bf v; uint4 q[2]; } u;
  u.q[0] = *(const uint4*)(p);
  u.q[1] = *(const uint4*)(p + 8);
  return u.v;
}

// B-fragment (32x16 bf16) from bf16 memory stored [n][k] row-major
// (i.e. B[k][n] = src[n][k]).  Lane layout: n = lane&15; lanes 0-15 hold
// K=0..15, lanes 16-31 hold K=16..31 (one 32B contiguous K-run per lane).
__device__ __forceinline__ v16bf load_b_bf16(const bf16_t* base, int stride) {
  const int lane = threadIdx.x & 31;
  const bf16_t* p = base + (size_t)(lane & 15) * stride + ((lane >> 4) << 4);
  union { v16bf v; uint4 q[2]; } u;
  u.q[0] = *(const uint4*)(p);
  u.q[1] = *(const uint4*)(p + 8);
  return u.v;
}

// A-fragment direct from global f32 [rows][stride], converting to bf16.
__device__ __forceinline__ v16bf gload_a_f32(const float* base, int stride) {
  const int lane = threadIdx.x & 31;
  const float* p = base + (size_t)(lane & 15) * stride + ((lane >> 4) << 4);
  v16bf a;
#pragma unroll
  for (int j = 0; j < 8; ++j) a[j] = (bf16_t)p[j];
#pragma unroll
  for (int j = 0; j < 8; ++j) a[8 + j] = (bf16_t)p[8 + j];
  return a;
}

// B-fragment direct from global f32 stored [n][stride] (B[k][n] = src[n][k]).
__device__ __forceinline__ v16bf gload_b_f32(const float* base, int stride) {
  const int lane = threadIdx.x & 31;
  const float* p = base + (size_t)(lane & 15) * stride + ((lane >> 4) << 4);
  v16bf b;
#pragma unroll
  for (int j = 0; j < 16; ++j) b[j] = (bf16_t)p[j];
  return b;
}

// Rank-16 LoRA A-fragment: K=16 real data (t[rows][16]), K=16..31 zero.
__device__ __forceinline__ v16bf gload_a_rank16(const float* t, int row0) {
  const int lane = threadIdx.x & 31;
  const float* p = t + (size_t)(row0 + (lane & 15)) * 16;
  v16bf a;
#pragma unroll
  for (int j = 0; j < 16; ++j) a[j] = (lane < 16) ? (bf16_t)p[j] : (bf16_t)0.0f;
  return a;
}

// Rank-16 LoRA B-fragment from BL[Nout][16]: lanes>=16 (K=16..31) are zero.
__device__ __forceinline__ v16bf gload_b_rank16(const float* BL, int col0) {
  const int lane = threadIdx.x & 31;
  v16bf b;
  if (lane < 16) {
    const float* p = BL + (size_t)(col0 + lane) * 16;
#pragma unroll
    for (int j = 0; j < 16; ++j) b[j] = (bf16_t)p[j];
  } else {
#pragma unroll
    for (int j = 0; j < 16; ++j) b[j] = (bf16_t)0.0f;
  }
  return b;
}

// Convert 16 contiguous f32 -> 16 contiguous bf16 (two uint4 stores).
__device__ __forceinline__ void stage16(const float* __restrict__ src,
                                        bf16_t* __restrict__ dst) {
  const float4* s = (const float4*)src;
  float4 f0 = s[0], f1 = s[1], f2 = s[2], f3 = s[3];
  union { uint4 q[2]; bf16_t h[16]; } u;
  u.h[0]=(bf16_t)f0.x; u.h[1]=(bf16_t)f0.y; u.h[2]=(bf16_t)f0.z; u.h[3]=(bf16_t)f0.w;
  u.h[4]=(bf16_t)f1.x; u.h[5]=(bf16_t)f1.y; u.h[6]=(bf16_t)f1.z; u.h[7]=(bf16_t)f1.w;
  u.h[8]=(bf16_t)f2.x; u.h[9]=(bf16_t)f2.y; u.h[10]=(bf16_t)f2.z; u.h[11]=(bf16_t)f2.w;
  u.h[12]=(bf16_t)f3.x; u.h[13]=(bf16_t)f3.y; u.h[14]=(bf16_t)f3.z; u.h[15]=(bf16_t)f3.w;
  *(uint4*)dst = u.q[0];
  *(uint4*)(dst + 8) = u.q[1];
}

// ---------------------------------------------------------------------------
// CDNA5 async global->LDS copy (ASYNCcnt-tracked).  vdst = per-lane LDS byte
// address, vaddr = 64-bit global address.  Completion via s_wait_asynccnt.
// ---------------------------------------------------------------------------
__device__ __forceinline__ unsigned lds_off(const void* p) {
  return (unsigned)(size_t)p;  // generic LDS address: low 32 bits = LDS offset
}
__device__ __forceinline__ void async_copy_b128(unsigned lds_byte,
                                                const void* gaddr) {
  asm volatile("global_load_async_to_lds_b128 %0, %1, off"
               :: "v"(lds_byte), "v"(gaddr) : "memory");
}
__device__ __forceinline__ void wait_async0() {
  asm volatile("s_wait_asynccnt 0" ::: "memory");
}

// ===========================================================================
// Kernel 1: t[4096][16] = 2.0 * (X[4096][K] @ A[16][K]^T)   (LoRA scale folded)
// One wave per 16-row tile. 8 waves per block, 32 blocks.
// ===========================================================================
__global__ __launch_bounds__(256) void lora_t_kernel(
    const float* __restrict__ X, const float* __restrict__ A,
    float* __restrict__ T, int K) {
  const int wave = blockIdx.x * 8 + (threadIdx.x >> 5);
  const int row0 = wave * 16;
  const int lane = threadIdx.x & 31;

  v8f acc = {0.f,0.f,0.f,0.f,0.f,0.f,0.f,0.f};
  for (int k0 = 0; k0 < K; k0 += 32) {
    v16bf a = gload_a_f32(X + (size_t)row0 * K + k0, K);
    v16bf b = gload_b_f32(A + k0, K);
    acc = wmma_bf16(a, b, acc);
  }
  const int col = lane & 15;
  const int rb  = row0 + ((lane >> 4) << 3);
#pragma unroll
  for (int v = 0; v < 8; ++v)
    T[(size_t)(rb + v) * 16 + col] = acc[v] * 2.0f;  // LORA_SCALE = 32/16
}

// ===========================================================================
// Kernel 2: GEMM + LoRA + bias.  Y = X @ W^T + T @ BL^T + bias.
// mode 0: store f32 to Yf[4096][Nout]              (projection, Nout=1024)
// mode 1: QKV: cols < 2048 (q,k) -> bf16 qkb[4096][2048];
//         cols >= 2048 (v)      -> bf16 vtb[B*H][64][2048] (transposed)
// 128x128 tile / 256-thread WG; 8 waves 4(M)x2(N); K staged 64-deep in LDS.
// ===========================================================================
__global__ __launch_bounds__(256) void gemm_lora_kernel(
    const float* __restrict__ X, const float* __restrict__ W,
    const float* __restrict__ T, const float* __restrict__ BL,
    const float* __restrict__ bias, float* __restrict__ Yf,
    bf16_t* __restrict__ qkb, bf16_t* __restrict__ vtb,
    int K, int Nout, int mode) {
  __shared__ bf16_t Xs[128 * 64];
  __shared__ bf16_t Ws[128 * 64];

  const int tid  = threadIdx.x;
  const int lane = tid & 31;
  const int row0 = blockIdx.y * 128;
  const int col0 = blockIdx.x * 128;
  const int w    = tid >> 5;
  const int wm   = (w & 3) * 32;   // wave row offset within tile
  const int wn   = (w >> 2) * 64;  // wave col offset within tile

  v8f acc[2][4];
#pragma unroll
  for (int i = 0; i < 2; ++i)
#pragma unroll
    for (int j = 0; j < 4; ++j)
      acc[i][j] = (v8f){0.f,0.f,0.f,0.f,0.f,0.f,0.f,0.f};

  const int sr = tid >> 1;         // staging row 0..127
  const int sc = (tid & 1) * 16;   // staging k-offset 0/16 (plus +32 copies)

  for (int k0 = 0; k0 < K; k0 += 64) {
    __syncthreads();
    stage16(X + (size_t)(row0 + sr) * K + k0 + sc,      Xs + sr * 64 + sc);
    stage16(X + (size_t)(row0 + sr) * K + k0 + sc + 32, Xs + sr * 64 + sc + 32);
    stage16(W + (size_t)(col0 + sr) * K + k0 + sc,      Ws + sr * 64 + sc);
    stage16(W + (size_t)(col0 + sr) * K + k0 + sc + 32, Ws + sr * 64 + sc + 32);
    if (k0 + 64 < K) {  // hint next tiles into cache (global_prefetch_b8)
      __builtin_prefetch(X + (size_t)(row0 + sr) * K + k0 + 64 + sc, 0, 1);
      __builtin_prefetch(W + (size_t)(col0 + sr) * K + k0 + 64 + sc, 0, 1);
    }
    __syncthreads();

#pragma unroll
    for (int kc = 0; kc < 2; ++kc) {
      v16bf a0 = load_a_bf16(Xs + (wm +  0) * 64 + kc * 32, 64);
      v16bf a1 = load_a_bf16(Xs + (wm + 16) * 64 + kc * 32, 64);
      v16bf b[4];
#pragma unroll
      for (int j = 0; j < 4; ++j)
        b[j] = load_b_bf16(Ws + (wn + j * 16) * 64 + kc * 32, 64);
#pragma unroll
      for (int j = 0; j < 4; ++j) {
        acc[0][j] = wmma_bf16(a0, b[j], acc[0][j]);
        acc[1][j] = wmma_bf16(a1, b[j], acc[1][j]);
      }
    }
  }

  // Rank-16 LoRA epilogue: one WMMA per tile (K padded 16 -> 32 with zeros)
  v16bf ta0 = gload_a_rank16(T, row0 + wm);
  v16bf ta1 = gload_a_rank16(T, row0 + wm + 16);
#pragma unroll
  for (int j = 0; j < 4; ++j) {
    v16bf lb = gload_b_rank16(BL, col0 + wn + j * 16);
    acc[0][j] = wmma_bf16(ta0, lb, acc[0][j]);
    acc[1][j] = wmma_bf16(ta1, lb, acc[1][j]);
  }

  // Bias + store.  C layout: row = v + 8*(lane>=16), col = lane&15.
#pragma unroll
  for (int j = 0; j < 4; ++j) {
    const int gcol = col0 + wn + j * 16 + (lane & 15);
    const float bv = bias[gcol];
#pragma unroll
    for (int i = 0; i < 2; ++i) {
      const int grow = row0 + wm + i * 16 + ((lane >> 4) << 3);
      if (mode == 0) {
#pragma unroll
        for (int v = 0; v < 8; ++v)
          Yf[(size_t)(grow + v) * Nout + gcol] = acc[i][j][v] + bv;
      } else if (gcol < 2048) {  // q,k -> bf16 row-major [4096][2048]
#pragma unroll
        for (int v = 0; v < 8; ++v)
          qkb[(size_t)(grow + v) * 2048 + gcol] = (bf16_t)(acc[i][j][v] + bv);
      } else {                   // v -> bf16 transposed [(b*16+h)][d][2048]
        const int hh = (gcol - 2048) >> 6;
        const int dd = (gcol - 2048) & 63;
        const int bb = grow >> 11;        // batch
        const int nn = grow & 2047;       // seq pos
        bf16_t* vrow = vtb + ((size_t)(bb * 16 + hh) * 64 + dd) * 2048;
#pragma unroll
        for (int v = 0; v < 8; ++v)
          vrow[nn + v] = (bf16_t)(acc[i][j][v] + bv);
      }
    }
  }
}

// ===========================================================================
// Kernel 3: flash attention (bf16 operands, f32 accum/softmax).
// grid = (N/128, H, B), 256 threads.  Wave w owns q rows [blk*128+w*16, +16).
// K/V tiles (64 keys) land in LDS via global_load_async_to_lds_b128.
// ===========================================================================
__global__ __launch_bounds__(256) void attn_kernel(
    const bf16_t* __restrict__ QK, const bf16_t* __restrict__ VT,
    const int* __restrict__ mask, float* __restrict__ O) {
  constexpr int Nn = 2048, Cc = 1024, KB = 64;
  const int b = blockIdx.z, h = blockIdx.y;
  const int tid = threadIdx.x, w = tid >> 5, lane = tid & 31;
  const int qrow0 = blockIdx.x * 128 + w * 16;

  __shared__ bf16_t Ks[KB * 72];        // [key][d],  stride 72 halves (pad 8)
  __shared__ bf16_t Vt[64 * 72];        // [d][key],  stride 72 halves (pad 8)
  __shared__ bf16_t Ps[8][16 * KB];     // per-wave P staging
  __shared__ int    Ms[KB];

  // Q fragments: 16 rows x 64 d -> two bf16 A-frags, straight from global.
  const bf16_t* qb = QK + (size_t)(b * Nn + qrow0) * 2048 + h * 64;
  v16bf qa0 = load_a_bf16(qb,      2048);
  v16bf qa1 = load_a_bf16(qb + 32, 2048);

  float m[8], l[8];
  v8f oacc[4];
#pragma unroll
  for (int v = 0; v < 8; ++v) { m[v] = -3.0e38f; l[v] = 0.f; }
#pragma unroll
  for (int nt = 0; nt < 4; ++nt)
    oacc[nt] = (v8f){0.f,0.f,0.f,0.f,0.f,0.f,0.f,0.f};

  const unsigned ks_base = lds_off(&Ks[0]);
  const unsigned vt_base = lds_off(&Vt[0]);
  const bf16_t* krow0 = QK + 1024 + h * 64;                         // k cols
  const bf16_t* vrow0 = VT + (size_t)(b * 16 + h) * 64 * 2048;      // v rows

  for (int kb0 = 0; kb0 < Nn; kb0 += KB) {
    __syncthreads();
    // Async-stage K tile (64 keys x 128B) and V^T tile (64 d x 128B):
    // 512 16B chunks each, 2 per thread, per-lane LDS addresses keep padding.
#pragma unroll
    for (int u = 0; u < 2; ++u) {
      const int cidx = tid * 2 + u;       // 0..511
      const int row  = cidx >> 3;         // 0..63
      const int off  = (cidx & 7) * 16;   // byte offset within 128B row
      const char* gk = (const char*)(krow0 + (size_t)(b * Nn + kb0 + row) * 2048) + off;
      async_copy_b128(ks_base + row * 144 + off, gk);
      const char* gv = (const char*)(vrow0 + (size_t)row * 2048 + kb0) + off;
      async_copy_b128(vt_base + row * 144 + off, gv);
    }
    if (tid < KB) Ms[tid] = mask[b * Nn + kb0 + tid];
    wait_async0();
    __syncthreads();

    // ---- S = Q @ K^T (4 tiles of 16x16, K-dim = d = 64 = 2 chunks) ----
    v8f s[4];
#pragma unroll
    for (int t = 0; t < 4; ++t) {
      v8f z = (v8f){0.f,0.f,0.f,0.f,0.f,0.f,0.f,0.f};
      z    = wmma_bf16(qa0, load_b_bf16(Ks + t * 16 * 72 +  0, 72), z);
      s[t] = wmma_bf16(qa1, load_b_bf16(Ks + t * 16 * 72 + 32, 72), z);
    }
    // ---- scale + mask (scale on f32 accumulators: exact) ----
#pragma unroll
    for (int t = 0; t < 4; ++t) {
      const bool keep = Ms[t * 16 + (lane & 15)] != 0;
#pragma unroll
      for (int v = 0; v < 8; ++v)
        s[t][v] = keep ? s[t][v] * 0.125f : -1.0e30f;
    }
    // ---- row max (across 4 tiles, then across the 16-lane half) ----
    float mb[8];
#pragma unroll
    for (int v = 0; v < 8; ++v) {
      float mm = s[0][v];
#pragma unroll
      for (int t = 1; t < 4; ++t) mm = fmaxf(mm, s[t][v]);
      mb[v] = mm;
    }
#pragma unroll
    for (int off = 1; off < 16; off <<= 1)
#pragma unroll
      for (int v = 0; v < 8; ++v)
        mb[v] = fmaxf(mb[v], __shfl_xor(mb[v], off, 32));
    // ---- online softmax update ----
    float corr[8], lb[8];
#pragma unroll
    for (int v = 0; v < 8; ++v) {
      const float mn = fmaxf(m[v], mb[v]);
      corr[v] = __expf(m[v] - mn);
      m[v] = mn;
      lb[v] = 0.f;
    }
#pragma unroll
    for (int t = 0; t < 4; ++t)
#pragma unroll
      for (int v = 0; v < 8; ++v) {
        const float p = __expf(s[t][v] - m[v]);
        s[t][v] = p;
        lb[v] += p;
      }
#pragma unroll
    for (int off = 1; off < 16; off <<= 1)
#pragma unroll
      for (int v = 0; v < 8; ++v)
        lb[v] += __shfl_xor(lb[v], off, 32);
#pragma unroll
    for (int v = 0; v < 8; ++v) l[v] = l[v] * corr[v] + lb[v];
#pragma unroll
    for (int nt = 0; nt < 4; ++nt)
#pragma unroll
      for (int v = 0; v < 8; ++v)
        oacc[nt][v] *= corr[v];

    // ---- P -> per-wave LDS (C layout -> A layout refragmentation) ----
    bf16_t* pw = &Ps[w][0];
#pragma unroll
    for (int t = 0; t < 4; ++t)
#pragma unroll
      for (int v = 0; v < 8; ++v)
        pw[(v + ((lane >> 4) << 3)) * KB + t * 16 + (lane & 15)] =
            (bf16_t)s[t][v];

    // ---- O += P @ V ----
    v16bf pa0 = load_a_bf16(pw,      KB);
    v16bf pa1 = load_a_bf16(pw + 32, KB);
#pragma unroll
    for (int nt = 0; nt < 4; ++nt) {
      oacc[nt] = wmma_bf16(pa0, load_b_bf16(Vt + nt * 16 * 72 +  0, 72), oacc[nt]);
      oacc[nt] = wmma_bf16(pa1, load_b_bf16(Vt + nt * 16 * 72 + 32, 72), oacc[nt]);
    }
  }

  // ---- normalize and store (f32, layout (B, N, H*d)) ----
  float inv[8];
#pragma unroll
  for (int v = 0; v < 8; ++v) inv[v] = (l[v] > 0.f) ? 1.0f / l[v] : 0.f;
  float* ob = O + (size_t)(b * Nn + qrow0) * Cc + h * 64;
#pragma unroll
  for (int nt = 0; nt < 4; ++nt)
#pragma unroll
    for (int v = 0; v < 8; ++v)
      ob[(size_t)(v + ((lane >> 4) << 3)) * Cc + nt * 16 + (lane & 15)] =
          oacc[nt][v] * inv[v];
}

// ===========================================================================
extern "C" void kernel_launch(void* const* d_in, const int* in_sizes, int n_in,
                              void* d_out, int out_size, void* d_ws,
                              size_t ws_size, hipStream_t stream) {
  (void)in_sizes; (void)n_in; (void)out_size; (void)ws_size;
  const float* x    = (const float*)d_in[0];
  const int*   msk  = (const int*)d_in[1];
  const float* Wqkv = (const float*)d_in[2];
  const float* bqkv = (const float*)d_in[3];
  const float* Aqkv = (const float*)d_in[4];
  const float* Bqkv = (const float*)d_in[5];
  const float* Wp   = (const float*)d_in[6];
  const float* bp   = (const float*)d_in[7];
  const float* Ap   = (const float*)d_in[8];
  const float* Bp   = (const float*)d_in[9];
  float* out = (float*)d_out;

  bf16_t* qkb = (bf16_t*)d_ws;                         // 4096 x 2048 bf16
  bf16_t* vtb = qkb + (size_t)4096 * 2048;             // 32 x 64 x 2048 bf16
  float* attnout = (float*)(vtb + (size_t)32 * 64 * 2048);  // 4096 x 1024 f32
  float* t1 = attnout + (size_t)4096 * 1024;           // 4096 x 16 f32
  float* t2 = t1 + (size_t)4096 * 16;                  // 4096 x 16 f32

  // 1) t1 = 2 * x @ Aqkv^T
  lora_t_kernel<<<32, 256, 0, stream>>>(x, Aqkv, t1, 1024);
  // 2) qkv = x @ Wqkv^T + t1 @ Bqkv^T + bqkv  -> bf16 (q,k row-major; v^T)
  gemm_lora_kernel<<<dim3(3072 / 128, 4096 / 128), 256, 0, stream>>>(
      x, Wqkv, t1, Bqkv, bqkv, nullptr, qkb, vtb, 1024, 3072, 1);
  // 3) flash attention
  attn_kernel<<<dim3(2048 / 128, 16, 2), 256, 0, stream>>>(qkb, vtb, msk,
                                                           attnout);
  // 4) t2 = 2 * attnout @ Ap^T
  lora_t_kernel<<<32, 256, 0, stream>>>(attnout, Ap, t2, 1024);
  // 5) out = attnout @ Wp^T + t2 @ Bp^T + bp  (f32)
  gemm_lora_kernel<<<dim3(1024 / 128, 4096 / 128), 256, 0, stream>>>(
      attnout, Wp, t2, Bp, bp, out, nullptr, nullptr, 1024, 1024, 0);
}
